// Attention_64768106823895
// MI455X (gfx1250) — compile-verified, matrix operand order checked by reading, not run on hardware
//
#include <hip/hip_runtime.h>

#define SEQ   4096
#define DIM   256
#define HEADS 8
#define HDIM  32
#define SCALE 0.0625f   /* 256^-0.5 */

typedef _Float16 v16h __attribute__((ext_vector_type(16)));
typedef _Float16 v8h  __attribute__((ext_vector_type(8)));
typedef float    v8f  __attribute__((ext_vector_type(8)));

union Frag16 { v16h v; uint4 u[2]; };
union H8     { v8h  v; uint4 u;    };

// Per-lane A/B fragment for v_wmma_f32_16x16x32_f16, loading a 32-half row.
// ISA 16-bit A layout: lane half g=L/16, halves h<8 -> k=h+8g, h>=8 -> k=h+8+8g
// i.e. two contiguous 8-half (16B) runs at offsets 8g and 16+8g.
__device__ __forceinline__ v16h load_frag(const _Float16* row, int g) {
  Frag16 f;
  f.u[0] = *(const uint4*)(row + 8 * g);
  f.u[1] = *(const uint4*)(row + 16 + 8 * g);
  return f.v;
}

__device__ __forceinline__ v8f wmma_f16(v16h a, v16h b, v8f c) {
  return __builtin_amdgcn_wmma_f32_16x16x32_f16(false, a, false, b, (short)0, c,
                                                false, false);
}

// Async global->LDS copy, 16B per lane, tracked by ASYNCcnt.
// vdst = per-lane LDS byte offset, vaddr = per-lane 64-bit global address.
__device__ __forceinline__ void async_ld16(unsigned lds_byte, unsigned long long gaddr) {
  asm volatile("global_load_async_to_lds_b128 %0, %1, off"
               :: "v"(lds_byte), "v"(gaddr) : "memory");
}
// Low 32 bits of a generic shared-aperture pointer == LDS byte offset.
__device__ __forceinline__ unsigned lds_off(const void* p) {
  return (unsigned)(unsigned long long)(uintptr_t)p;
}

// ---------------- conversion / transpose prep ----------------
__global__ void k_cvt_x(const float* __restrict__ x, _Float16* __restrict__ xh) {
  int i = blockIdx.x * 256 + threadIdx.x;          // 4096*256 elems exact
  xh[i] = (_Float16)x[i];
}
__global__ void k_cvt_wq(const float* __restrict__ w, _Float16* __restrict__ wqT) {
  int i = blockIdx.x * 256 + threadIdx.x;          // 256*768 elems exact
  int k = i / 768, n = i % 768;
  wqT[n * 256 + k] = (_Float16)w[i];
}
__global__ void k_cvt_wo(const float* __restrict__ w, _Float16* __restrict__ woT) {
  int i = blockIdx.x * 256 + threadIdx.x;          // 256*256 elems exact
  int k = i >> 8, n = i & 255;
  woT[n * 256 + k] = (_Float16)w[i];
}

// ---------------- QKV projection: [4096,256]x[256,768] ----------------
__global__ __launch_bounds__(128) void k_qkv(const _Float16* __restrict__ xh,
                                             const _Float16* __restrict__ wqT,
                                             const float* __restrict__ bqkv,
                                             _Float16* __restrict__ Qh,
                                             _Float16* __restrict__ Kh,
                                             _Float16* __restrict__ Vth) {
  int t    = blockIdx.x * 4 + (threadIdx.x >> 5);  // 12288 tiles
  int mt   = t & 255;                              // 256 M tiles
  int nt   = t >> 8;                               // 48  N tiles
  int lane = threadIdx.x & 31;
  int n = lane & 15, g = lane >> 4;

  const _Float16* arow = xh  + (mt * 16 + n) * 256;
  const _Float16* brow = wqT + (nt * 16 + n) * 256;
  v8f c = {0.f, 0.f, 0.f, 0.f, 0.f, 0.f, 0.f, 0.f};
#pragma unroll
  for (int kc = 0; kc < 8; ++kc) {
    v16h a = load_frag(arow + kc * 32, g);
    v16h b = load_frag(brow + kc * 32, g);
    c = wmma_f16(a, b, c);
  }
  int col  = nt * 16 + n;
  float bias = bqkv[col];
  int sec  = col >> 8;          // 0=Q 1=K 2=V (uniform per wave)
  int cc   = col & 255;
  int head = cc >> 5, dim = cc & 31;
  int row0 = mt * 16 + 8 * g;   // C elem r -> row row0+r
  if (sec == 0) {
    _Float16* qp = Qh + ((head * SEQ) + row0) * HDIM + dim;
#pragma unroll
    for (int r = 0; r < 8; ++r) qp[r * HDIM] = (_Float16)((c[r] + bias) * SCALE);
  } else if (sec == 1) {
    _Float16* kp = Kh + ((head * SEQ) + row0) * HDIM + dim;
#pragma unroll
    for (int r = 0; r < 8; ++r) kp[r * HDIM] = (_Float16)(c[r] + bias);
  } else {
    H8 hv;
#pragma unroll
    for (int r = 0; r < 8; ++r) hv.v[r] = (_Float16)(c[r] + bias);
    *(uint4*)(Vth + (head * HDIM + dim) * SEQ + row0) = hv.u;   // contiguous rows
  }
}

// ---------------- flash attention (transposed-S, async double-buffered) ----------------
__global__ __launch_bounds__(128) void k_attn(const _Float16* __restrict__ Qh,
                                              const _Float16* __restrict__ Kh,
                                              const _Float16* __restrict__ Vth,
                                              _Float16* __restrict__ Oh) {
  __shared__ __align__(16) _Float16 Kt[2][64 * 32];   // [buf][seq][dim]
  __shared__ __align__(16) _Float16 Vt[2][32 * 64];   // [buf][dim][seq]
  const int ib = blockIdx.x, hd = blockIdx.y;
  const int tid  = threadIdx.x;
  const int lane = tid & 31, wave = tid >> 5;
  const int n = lane & 15, g = lane >> 4;
  const int qrow = ib * 64 + wave * 16 + n;

  const _Float16* kbase = Kh  + hd * SEQ * HDIM;
  const _Float16* vbase = Vth + hd * HDIM * SEQ;

  // Issue one tile's async global->LDS copies (4 async instrs per wave).
  auto stage = [&](int buf, int jb) {
    const _Float16* ksrc = kbase + jb * 64 * HDIM;   // contiguous 4KB
#pragma unroll
    for (int ii = 0; ii < 2; ++ii) {
      int i = tid + ii * 128;                        // 256 x 16B for K
      async_ld16(lds_off(&Kt[buf][0]) + i * 16,
                 (unsigned long long)(uintptr_t)(ksrc) + i * 16);
    }
#pragma unroll
    for (int ii = 0; ii < 2; ++ii) {
      int i = tid + ii * 128;                        // 256 x 16B for V^T
      int d = i >> 3, q = i & 7;
      async_ld16(lds_off(&Vt[buf][0]) + i * 16,
                 (unsigned long long)(uintptr_t)(vbase + d * SEQ + jb * 64 + q * 8));
    }
  };

  // Q rows as B-operand fragment (Q already pre-scaled by SCALE)
  v16h bq = load_frag(Qh + (hd * SEQ + qrow) * HDIM, g);

  float m = -__builtin_inff(), l = 0.f;
  v8f o0 = {0.f, 0.f, 0.f, 0.f, 0.f, 0.f, 0.f, 0.f};
  v8f o1 = {0.f, 0.f, 0.f, 0.f, 0.f, 0.f, 0.f, 0.f};

  stage(0, 0);
  for (int jb = 0; jb < SEQ / 64; ++jb) {
    const int cur = jb & 1;
    __syncthreads();                        // prior compute done before overwrite
    if (jb + 1 < SEQ / 64) {
      stage(cur ^ 1, jb + 1);               // prefetch next tile
      asm volatile("s_wait_asynccnt 4" ::: "memory");   // 4 older ops (cur) done
    } else {
      asm volatile("s_wait_asynccnt 0" ::: "memory");
    }
    __syncthreads();                        // all waves' cur tile resident

    // S^T = K_sub x Q^T : elem r of st[jt] = S[qrow][jb*64 + jt*16 + r + 8g]
    v8f st[4];
#pragma unroll
    for (int jt = 0; jt < 4; ++jt) {
      v16h ak = load_frag(&Kt[cur][(jt * 16 + n) * HDIM], g);
      v8f z = {0.f, 0.f, 0.f, 0.f, 0.f, 0.f, 0.f, 0.f};
      st[jt] = wmma_f16(ak, bq, z);
    }

    // online softmax (row == qrow == lane%16 -> scalar stats per lane)
    float smax = -__builtin_inff();
#pragma unroll
    for (int f = 0; f < 4; ++f)
#pragma unroll
      for (int r = 0; r < 8; ++r) smax = fmaxf(smax, st[f][r]);
    smax = fmaxf(smax, __shfl_xor(smax, 16, 32));
    float mn    = fmaxf(m, smax);
    float alpha = __expf(m - mn);

    // P^T fragments: pb[c] halves 0..7 <- st[2c], halves 8..15 <- st[2c+1]
    v16h pb[2];
    float rs = 0.f;
#pragma unroll
    for (int cch = 0; cch < 2; ++cch)
#pragma unroll
      for (int h = 0; h < 16; ++h) {
        float p = __expf(st[2 * cch + (h >> 3)][h & 7] - mn);
        rs += p;
        pb[cch][h] = (_Float16)p;
      }
    rs += __shfl_xor(rs, 16, 32);
    l = l * alpha + rs;
#pragma unroll
    for (int r = 0; r < 8; ++r) { o0[r] *= alpha; o1[r] *= alpha; }

    // O^T += V^T_chunk x P_chunk
#pragma unroll
    for (int cch = 0; cch < 2; ++cch) {
      v16h av0 = load_frag(&Vt[cur][n * 64 + cch * 32], g);
      o0 = wmma_f16(av0, pb[cch], o0);
      v16h av1 = load_frag(&Vt[cur][(16 + n) * 64 + cch * 32], g);
      o1 = wmma_f16(av1, pb[cch], o1);
    }
    m = mn;
  }

  float inv = 1.f / l;
  H8 h0, h1;
#pragma unroll
  for (int r = 0; r < 8; ++r) {
    h0.v[r] = (_Float16)(o0[r] * inv);   // dims r+8g
    h1.v[r] = (_Float16)(o1[r] * inv);   // dims 16+r+8g
  }
  _Float16* orow = Oh + qrow * DIM + hd * HDIM;
  *(uint4*)(orow + 8 * g)      = h0.u;
  *(uint4*)(orow + 16 + 8 * g) = h1.u;
}

// ---------------- output projection: [4096,256]x[256,256] + bias ----------------
__global__ __launch_bounds__(128) void k_proj(const _Float16* __restrict__ Oh,
                                              const _Float16* __restrict__ woT,
                                              const float* __restrict__ bo,
                                              float* __restrict__ out) {
  int t    = blockIdx.x * 4 + (threadIdx.x >> 5);  // 4096 tiles
  int ntl  = t & 15;
  int mt   = t >> 4;
  int lane = threadIdx.x & 31;
  int n = lane & 15, g = lane >> 4;

  const _Float16* arow = Oh  + (mt * 16 + n) * 256;
  const _Float16* brow = woT + (ntl * 16 + n) * 256;
  v8f c = {0.f, 0.f, 0.f, 0.f, 0.f, 0.f, 0.f, 0.f};
#pragma unroll
  for (int kc = 0; kc < 8; ++kc) {
    v16h a = load_frag(arow + kc * 32, g);
    v16h b = load_frag(brow + kc * 32, g);
    c = wmma_f16(a, b, c);
  }
  int col = ntl * 16 + n;
  float bias = bo[col];
  int row0 = mt * 16 + 8 * g;
#pragma unroll
  for (int r = 0; r < 8; ++r) out[(row0 + r) * DIM + col] = c[r] + bias;
}

extern "C" void kernel_launch(void* const* d_in, const int* in_sizes, int n_in,
                              void* d_out, int out_size, void* d_ws, size_t ws_size,
                              hipStream_t stream) {
  (void)in_sizes; (void)n_in; (void)out_size; (void)ws_size;
  const float* x   = (const float*)d_in[0];
  const float* wq  = (const float*)d_in[1];
  const float* bq  = (const float*)d_in[2];
  const float* wo  = (const float*)d_in[3];
  const float* bo  = (const float*)d_in[4];
  float* out = (float*)d_out;

  char* ws = (char*)d_ws;
  _Float16* xh  = (_Float16*)(ws + 0);          // 2 MB
  _Float16* wqT = (_Float16*)(ws + 2097152);    // 384 KB  (wqT[n][k])
  _Float16* woT = (_Float16*)(ws + 2490368);    // 128 KB  (woT[n][k])
  _Float16* Qh  = (_Float16*)(ws + 2621440);    // 2 MB  [h][seq][d], pre-scaled
  _Float16* Kh  = (_Float16*)(ws + 4718592);    // 2 MB  [h][seq][d]
  _Float16* Vth = (_Float16*)(ws + 6815744);    // 2 MB  [h][d][seq]
  _Float16* Oh  = (_Float16*)(ws + 8912896);    // 2 MB  [seq][dim]

  k_cvt_x <<<4096, 256, 0, stream>>>(x, xh);
  k_cvt_wq<<<768,  256, 0, stream>>>(wq, wqT);
  k_cvt_wo<<<256,  256, 0, stream>>>(wo, woT);
  k_qkv   <<<3072, 128, 0, stream>>>(xh, wqT, bq, Qh, Kh, Vth);
  k_attn  <<<dim3(SEQ / 64, HEADS), 128, 0, stream>>>(Qh, Kh, Vth, Oh);
  k_proj  <<<1024, 128, 0, stream>>>(Oh, woT, bo, out);
}
